// Varkeys_16423954940627
// MI455X (gfx1250) — matrix operand order repaired and checked by reading, not compiled, not run on hardware
//
#include <hip/hip_runtime.h>
#include <hip/hip_bf16.h>

// ---------------------------------------------------------------------------
// Varkeys fused kernel for MI455X (gfx1250, wave32, WMMA bf16 + TDM async DMA)
//   S  = x . keys^T          (4096 x 16384, K=512)   -> fused, never stored
//   kern = 1e4 / (||k||^2 + ||x||^2 - 2S + 1)
//   KV = kern . V            (4096 x 128)
//   out = KV / rowsum(KV)
// ---------------------------------------------------------------------------

typedef __attribute__((ext_vector_type(16))) __bf16 v16bf;
typedef __attribute__((ext_vector_type(8)))  __bf16 v8bf;
typedef __attribute__((ext_vector_type(8)))  float  v8f;
typedef __attribute__((ext_vector_type(4)))  unsigned int u32x4;
typedef __attribute__((ext_vector_type(8)))  int i32x8;
typedef __attribute__((ext_vector_type(4)))  int i32x4;

#define B_DIM 4096
#define K_DIM 512
#define D_DIM 16384
#define C_DIM 128
#define BTILE 64
#define DCHUNK 64
#define DSPLIT 4
#define DRANGE (D_DIM / DSPLIT)
#define NCHUNK (DRANGE / DCHUNK)

// LDS pitches (bf16 elements); pitch*2 bytes mult of 16, bank-spreading pad.
#define XP 520
#define KP 520
#define GP 72
#define VP 72
// sX[64*XP] | sK0[64*KP] | sK1[64*KP] | sG[64*GP] | sV0[128*VP] | sV1[128*VP] | sXn[64]f32
#define SMEM_BYTES ((BTILE*XP + 2*DCHUNK*KP + BTILE*GP + 2*C_DIM*VP) * 2 + BTILE * 4)

// ------------------------- TDM descriptor helper ---------------------------
// Issues tensor_load_to_lds for a 2D tile of bf16 (data_size code 1):
//   tile_d0 elements per row, tile_d1 rows, row stride stride0 (elements),
//   LDS rows padded: after every 2^(pad_int+1) DWORDs add (pad_amt+1) DWORDs.
#if __has_include(<hip/amd_detail/amd_gfx1250_TDM.h>)
#define TDM_CALL(g0, g1, gz) \
    __builtin_amdgcn_tensor_load_to_lds(g0, g1, gz, gz, (i32x8){0,0,0,0,0,0,0,0}, 0)
#else
#define TDM_CALL(g0, g1, gz) \
    __builtin_amdgcn_tensor_load_to_lds(g0, g1, gz, gz, 0)
#endif

__device__ __forceinline__ void tdm_load_2d_bf16(
        unsigned lds_off, const void* gptr,
        unsigned tile_d0, unsigned tile_d1,
        unsigned tensor_d0, unsigned tensor_d1,
        unsigned long long stride0,
        unsigned pad_int, unsigned pad_amt) {
    unsigned long long ga = (unsigned long long)(size_t)gptr;
    u32x4 g0;
    g0[0] = 1u;                                            // count=1, user D#
    g0[1] = lds_off;                                       // lds_addr
    g0[2] = (unsigned)ga;                                  // global_addr[31:0]
    g0[3] = (unsigned)((ga >> 32) & 0x01FFFFFFu) | (2u << 30); // addr[56:32]|type=2
    i32x8 g1;
    g1[0] = (int)((1u << 16)                               // data_size = 2 bytes
                | (1u << 20)                               // pad_enable
                | (pad_int << 22) | (pad_amt << 25));
    g1[1] = (int)((tensor_d0 & 0xFFFFu) << 16);            // dim0[15:0] (bar addr 0)
    g1[2] = (int)(((tensor_d0 >> 16) & 0xFFFFu) | ((tensor_d1 & 0xFFFFu) << 16));
    g1[3] = (int)(((tensor_d1 >> 16) & 0xFFFFu) | ((tile_d0 & 0xFFFFu) << 16));
    g1[4] = (int)(tile_d1 & 0xFFFFu);                      // tile_dim1, tile_dim2=0
    g1[5] = (int)(unsigned)(stride0 & 0xFFFFFFFFu);        // dim0_stride lo
    g1[6] = (int)(unsigned)((stride0 >> 32) & 0xFFFFu);    // dim0_stride hi, dim1_stride=0
    g1[7] = 0;
    i32x4 gz = {0, 0, 0, 0};
    TDM_CALL(g0, g1, gz);
}

// ---------------- prep: f32 rows -> bf16 rows + squared norms --------------
__global__ void prep_rows(const float* __restrict__ src, __bf16* __restrict__ dst,
                          float* __restrict__ norms, int nrows, int rowlen) {
    int w    = (blockIdx.x * blockDim.x + threadIdx.x) >> 5;   // one wave32/row
    int lane = threadIdx.x & 31;
    if (w >= nrows) return;
    const float* r = src + (size_t)w * rowlen;
    __bf16*      o = dst + (size_t)w * rowlen;
    float acc = 0.f;
    for (int k = lane * 4; k < rowlen; k += 128) {
        float4 v = *(const float4*)(r + k);
        acc += v.x*v.x + v.y*v.y + v.z*v.z + v.w*v.w;
        o[k+0] = (__bf16)v.x; o[k+1] = (__bf16)v.y;
        o[k+2] = (__bf16)v.z; o[k+3] = (__bf16)v.w;
    }
    #pragma unroll
    for (int d = 16; d > 0; d >>= 1) acc += __shfl_down(acc, d);
    if (lane == 0) norms[w] = acc;
}

// ---------------- prep: V (D x C) -> Vt bf16 (C x D) -----------------------
__global__ void prep_vt(const float* __restrict__ V, __bf16* __restrict__ Vt) {
    int idx = blockIdx.x * blockDim.x + threadIdx.x;
    if (idx >= D_DIM * C_DIM) return;
    int d = idx & (D_DIM - 1);
    int c = idx >> 14;
    Vt[(size_t)c * D_DIM + d] = (__bf16)V[(size_t)d * C_DIM + c];
}

// ---------------- WMMA operand loaders (ISA 7.12.2 layouts) ----------------
__device__ __forceinline__ v16bf load_matA(const __bf16* base, int pitch,
                                           int row0, int k0, int lane) {
    int m  = row0 + (lane & 15);
    int kb = k0 + ((lane & 16) ? 8 : 0);
    const __bf16* p = base + m * pitch + kb;
    v8bf lo = *(const v8bf*)(p);
    v8bf hi = *(const v8bf*)(p + 16);
    v16bf a;
    #pragma unroll
    for (int i = 0; i < 8; ++i) { a[i] = lo[i]; a[i + 8] = hi[i]; }
    return a;
}
__device__ __forceinline__ v16bf load_matB(const __bf16* base, int pitch,
                                           int k0, int n0, int lane) {
    int n  = n0 + (lane & 15);
    int kb = k0 + ((lane & 16) ? 8 : 0);
    const __bf16* p = base + n * pitch + kb;
    v8bf lo = *(const v8bf*)(p);
    v8bf hi = *(const v8bf*)(p + 16);
    v16bf b;
    #pragma unroll
    for (int i = 0; i < 8; ++i) { b[i] = lo[i]; b[i + 8] = hi[i]; }
    return b;
}

// ---------------- fused main kernel ----------------------------------------
__global__ __launch_bounds__(256, 1)
void varkeys_main(const __bf16* __restrict__ xb,  const __bf16* __restrict__ kb,
                  const __bf16* __restrict__ vt,  const float* __restrict__ xnorm,
                  const float* __restrict__ knorm, float* __restrict__ part) {
    extern __shared__ char smem[];
    __bf16* sX  = (__bf16*)smem;                   // [64][XP]
    __bf16* sK0 = sX  + BTILE * XP;                // [64][KP] double-buffered
    __bf16* sK1 = sK0 + DCHUNK * KP;
    __bf16* sG  = sK1 + DCHUNK * KP;               // [64][GP]
    __bf16* sV0 = sG  + BTILE * GP;                // [128][VP] double-buffered
    __bf16* sV1 = sV0 + C_DIM * VP;
    float*  sXn = (float*)(sV1 + C_DIM * VP);      // [64]

    const int tid  = threadIdx.x;
    const int lane = tid & 31;
    const int w    = tid >> 5;
    const int rs   = w & 3;
    const int h    = w >> 2;
    const int brow0  = blockIdx.x * BTILE;
    const int dbase0 = blockIdx.y * DRANGE;

    const unsigned offX  = (unsigned)(size_t)(void*)sX;
    const unsigned offK0 = (unsigned)(size_t)(void*)sK0;
    const unsigned offK1 = (unsigned)(size_t)(void*)sK1;
    const unsigned offV0 = (unsigned)(size_t)(void*)sV0;
    const unsigned offV1 = (unsigned)(size_t)(void*)sV1;

    // prologue: async TDM of x tile + chunk0 keys/V (wave 0 issues, in-order)
    if (w == 0) {
        tdm_load_2d_bf16(offX, xb + (size_t)brow0 * K_DIM,
                         K_DIM, BTILE, K_DIM, BTILE, K_DIM, 7, 3);   // 1024B rows +16B pad
        tdm_load_2d_bf16(offK0, kb + (size_t)dbase0 * K_DIM,
                         K_DIM, DCHUNK, K_DIM, DCHUNK, K_DIM, 7, 3);
        tdm_load_2d_bf16(offV0, vt + dbase0,
                         DCHUNK, C_DIM, DCHUNK, C_DIM, D_DIM, 4, 3); // 128B rows +16B pad
    }
    if (tid < BTILE) sXn[tid] = xnorm[brow0 + tid];
    if (w == 0) __builtin_amdgcn_s_wait_tensorcnt(2);   // x tile resident
    __syncthreads();

    float xn[8];
    {
        const int mrow = rs * 16 + ((lane & 16) ? 8 : 0);
        #pragma unroll
        for (int v = 0; v < 8; ++v) xn[v] = sXn[mrow + v];
    }
    // x A-fragments are invariant across all D-chunks: keep resident in VGPRs
    v16bf afrag[K_DIM / 32];
    #pragma unroll
    for (int ks = 0; ks < K_DIM / 32; ++ks)
        afrag[ks] = load_matA(sX, XP, rs * 16, ks * 32, lane);

    v8f o[4] = {};

    for (int chunk = 0; chunk < NCHUNK; ++chunk) {
        const int dbase = dbase0 + chunk * DCHUNK;
        __bf16* sKb = (chunk & 1) ? sK1 : sK0;
        __bf16* sVb = (chunk & 1) ? sV1 : sV0;

        if (w == 0) __builtin_amdgcn_s_wait_tensorcnt(0); // chunk's DMA done
        __syncthreads();                                  // publish buf, free buf^1

        if (w == 0 && chunk + 1 < NCHUNK) {               // prefetch next chunk via TDM
            const int dnext = dbase + DCHUNK;
            tdm_load_2d_bf16((chunk & 1) ? offK0 : offK1,
                             kb + (size_t)dnext * K_DIM,
                             K_DIM, DCHUNK, K_DIM, DCHUNK, K_DIM, 7, 3);
            tdm_load_2d_bf16((chunk & 1) ? offV0 : offV1,
                             vt + dnext,
                             DCHUNK, C_DIM, DCHUNK, C_DIM, D_DIM, 4, 3);
        }

        // ---- GEMM1: S strip = x(16 x 512) . keys_chunk^T (512 x 32) ----
        v8f s0 = {}, s1 = {};
        #pragma unroll
        for (int ks = 0; ks < K_DIM / 32; ++ks) {
            v16bf b0 = load_matB(sKb, KP, ks * 32, h * 32,      lane);
            v16bf b1 = load_matB(sKb, KP, ks * 32, h * 32 + 16, lane);
            s0 = __builtin_amdgcn_wmma_f32_16x16x32_bf16(false, afrag[ks], false, b0,
                                                         (short)0, s0, false, false);
            s1 = __builtin_amdgcn_wmma_f32_16x16x32_bf16(false, afrag[ks], false, b1,
                                                         (short)0, s1, false, false);
        }

        // ---- rational kernel + write bf16 kern tile to LDS ----
        {
            const int dloc = h * 32 + (lane & 15);
            const float kn0 = knorm[dbase + dloc];
            const float kn1 = knorm[dbase + dloc + 16];
            const int mrow  = rs * 16 + ((lane & 16) ? 8 : 0);
            #pragma unroll
            for (int v = 0; v < 8; ++v) {
                float sq0 = kn0 + xn[v] - 2.f * s0[v];
                float sq1 = kn1 + xn[v] - 2.f * s1[v];
                float g0  = 10000.f / (sq0 + 1.0f);   // == 1/(sq/1e4 + 1e-4)
                float g1  = 10000.f / (sq1 + 1.0f);
                sG[(mrow + v) * GP + dloc]      = (__bf16)g0;
                sG[(mrow + v) * GP + dloc + 16] = (__bf16)g1;
            }
        }
        __syncthreads();

        // ---- GEMM2: O(16 x 64) += kern(16 x 64) . V_chunk(64 x 64-half) ----
        #pragma unroll
        for (int kk = 0; kk < DCHUNK; kk += 32) {
            v16bf a2 = load_matA(sG, GP, rs * 16, kk, lane);
            #pragma unroll
            for (int j = 0; j < 4; ++j) {
                v16bf bv = load_matB(sVb, VP, kk, h * 64 + j * 16, lane);
                o[j] = __builtin_amdgcn_wmma_f32_16x16x32_bf16(false, a2, false, bv,
                                                               (short)0, o[j], false, false);
            }
        }
    }

    // ---- write f32 partial KV tile ----
    {
        const int mrow = rs * 16 + ((lane & 16) ? 8 : 0);
        #pragma unroll
        for (int j = 0; j < 4; ++j) {
            const int cg = h * 64 + j * 16 + (lane & 15);
            #pragma unroll
            for (int v = 0; v < 8; ++v) {
                const int bg = brow0 + mrow + v;
                part[((size_t)blockIdx.y * B_DIM + bg) * C_DIM + cg] = o[j][v];
            }
        }
    }
}

// ---------------- finish: reduce D-splits + row-normalize ------------------
__global__ void finish(const float* __restrict__ part, float* __restrict__ out) {
    const int b = blockIdx.x;
    const int c = threadIdx.x;                      // C_DIM threads
    float v = 0.f;
    #pragma unroll
    for (int s = 0; s < DSPLIT; ++s)
        v += part[((size_t)s * B_DIM + b) * C_DIM + c];
    __shared__ float red[C_DIM];
    red[c] = v;
    __syncthreads();
    for (int off = C_DIM / 2; off > 0; off >>= 1) {
        if (c < off) red[c] += red[c + off];
        __syncthreads();
    }
    out[(size_t)b * C_DIM + c] = v / red[0];
}

// ---------------------------------------------------------------------------
extern "C" void kernel_launch(void* const* d_in, const int* in_sizes, int n_in,
                              void* d_out, int out_size, void* d_ws, size_t ws_size,
                              hipStream_t stream) {
    const float* x    = (const float*)d_in[0];   // (4096, 512)
    const float* keys = (const float*)d_in[1];   // (16384, 512)
    const float* V    = (const float*)d_in[2];   // (16384, 128)
    float*       out  = (float*)d_out;           // (4096, 128)

    char*  ws  = (char*)d_ws;
    size_t off = 0;
    auto carve = [&](size_t bytes) -> char* {
        char* p = ws + off;
        off = (off + bytes + 255) & ~(size_t)255;
        return p;
    };
    float*  xnorm = (float*)carve((size_t)B_DIM * 4);
    float*  knorm = (float*)carve((size_t)D_DIM * 4);
    __bf16* xb    = (__bf16*)carve((size_t)B_DIM * K_DIM * 2);
    __bf16* kbf   = (__bf16*)carve((size_t)D_DIM * K_DIM * 2);
    __bf16* vt    = (__bf16*)carve((size_t)C_DIM * D_DIM * 2);
    float*  part  = (float*)carve((size_t)DSPLIT * B_DIM * C_DIM * 4);

    prep_rows<<<dim3((B_DIM * 32) / 256), dim3(256), 0, stream>>>(x, xb, xnorm, B_DIM, K_DIM);
    prep_rows<<<dim3((D_DIM * 32) / 256), dim3(256), 0, stream>>>(keys, kbf, knorm, D_DIM, K_DIM);
    prep_vt  <<<dim3((D_DIM * C_DIM) / 256), dim3(256), 0, stream>>>(V, vt);

    varkeys_main<<<dim3(B_DIM / BTILE, DSPLIT), dim3(256), SMEM_BYTES, stream>>>(
        xb, kbf, vt, xnorm, knorm, part);

    finish<<<dim3(B_DIM), dim3(C_DIM), 0, stream>>>(part, out);
}
